// ImageModalityEncoder_62148176773397
// MI455X (gfx1250) — compile-verified, matrix operand order checked by reading, not run on hardware
//
#include <hip/hip_runtime.h>
#include <hip/hip_bf16.h>
#include <math.h>
#include <stddef.h>

// ---- problem constants ----
#define BB     8
#define TT     90
#define HH     290
#define WW     180
#define PSS    10
#define NWPP   18
#define NPP    522
#define PDD    100
#define NVV    94
#define KVIS   24
#define LLEN   2160            // T*K
#define DMOD   256
#define NHEADS 8
#define DHEAD  32
#define NLAY   6
#define DFFN   1024
#define MTOT   (BB*LLEN)       // 17280 rows, divisible by 128

typedef __attribute__((ext_vector_type(16))) __bf16 bf16x16;
typedef __attribute__((ext_vector_type(8)))  __bf16 bf16x8;
typedef __attribute__((ext_vector_type(8)))  float  f32x8;
typedef __attribute__((__vector_size__(4 * sizeof(int)))) int v4i;

static __device__ __forceinline__ __bf16 f2bf(float f) {
    unsigned u = __builtin_bit_cast(unsigned, f);
    unsigned r = (u + 0x7FFFu + ((u >> 16) & 1u)) >> 16;   // round-to-nearest-even
    unsigned short s = (unsigned short)r;
    return __builtin_bit_cast(__bf16, s);
}

// ---- CDNA5 async global->LDS path (ASYNCcnt), with portable fallback ----
#if defined(__HIP_DEVICE_COMPILE__) && __has_builtin(__builtin_amdgcn_global_load_async_to_lds_b128)
#define USE_ASYNC_LDS 1
#endif
#if defined(__HIP_DEVICE_COMPILE__) && __has_builtin(__builtin_amdgcn_s_wait_asynccnt)
#define HAVE_WAIT_ASYNC 1
#endif

#ifdef USE_ASYNC_LDS
typedef __attribute__((address_space(1))) v4i gv4i;   // global (__device__) AS
typedef __attribute__((address_space(3))) v4i lv4i;   // LDS (__shared__) AS
static __device__ __forceinline__ void async_cp16(void* lds, const void* g) {
    __builtin_amdgcn_global_load_async_to_lds_b128(
        (gv4i*)(g), (lv4i*)(lds), 0, 0);
}
static __device__ __forceinline__ void async_wait_last3() {
#ifdef HAVE_WAIT_ASYNC
    __builtin_amdgcn_s_wait_asynccnt(3);
#else
    asm volatile("s_wait_asynccnt 0x3" ::: "memory");
#endif
}
static __device__ __forceinline__ void async_wait_all() {
#ifdef HAVE_WAIT_ASYNC
    __builtin_amdgcn_s_wait_asynccnt(0);
#else
    asm volatile("s_wait_asynccnt 0x0" ::: "memory");
#endif
}
#endif

// ---------------------------------------------------------------------------
// Kernel 1: gather visible patches -> bf16 A rows (K padded 100->128) + meta
// ---------------------------------------------------------------------------
__global__ __launch_bounds__(128)
void build_tokens_kernel(const float* __restrict__ x_img,
                         const unsigned char* __restrict__ patch_mask,
                         const int* __restrict__ valid_idx,
                         __bf16* __restrict__ tokA,
                         int* __restrict__ meta) {
    int bt = blockIdx.x;                 // b*T + t
    int b  = bt / TT, t = bt % TT;
    __shared__ int vis[KVIS];
    if (threadIdx.x < KVIS) vis[threadIdx.x] = 0;
    __syncthreads();
    if (threadIdx.x == 0) {
        int cnt = 0;
        for (int nv = 0; nv < NVV && cnt < KVIS; ++nv) {
            int np = valid_idx[nv];
            if (!patch_mask[(size_t)bt * NPP + np]) vis[cnt++] = nv;
        }
    }
    __syncthreads();
    int tid = threadIdx.x;
    for (int j = 0; j < KVIS; ++j) {
        int nv = vis[j];
        if (nv < 0) nv = 0; if (nv >= NVV) nv = NVV - 1;
        size_t row = (size_t)b * LLEN + (size_t)t * KVIS + j;
        if (tid == 0) { meta[row * 2] = t; meta[row * 2 + 1] = nv; }
        int np = valid_idx[nv];
        int nh = np / NWPP, nw = np % NWPP;
        if (tid < PDD) {
            int r = tid / PSS, c = tid % PSS;
            float v = x_img[(((size_t)bt) * HH + nh * PSS + r) * WW + nw * PSS + c];
            tokA[row * 128 + tid] = f2bf(v);
        } else {
            tokA[row * 128 + tid] = f2bf(0.f);
        }
    }
}

// ---------------------------------------------------------------------------
// Kernel 2: fp32 -> bf16 bulk convert (weights, once)
// ---------------------------------------------------------------------------
__global__ __launch_bounds__(256)
void cvt_bf16_kernel(const float* __restrict__ in, __bf16* __restrict__ out, int n) {
    int i = blockIdx.x * 256 + threadIdx.x;
    if (i < n) out[i] = f2bf(in[i]);
}

// pad patch_embed_w (256x100) -> bf16 (256x128) with zeros
__global__ __launch_bounds__(256)
void pad_w_kernel(const float* __restrict__ w, __bf16* __restrict__ wpad) {
    int idx = blockIdx.x * 256 + threadIdx.x;   // 256*128 elements
    int n = idx >> 7, k = idx & 127;
    wpad[(size_t)n * 128 + k] = (k < PDD) ? f2bf(w[(size_t)n * PDD + k]) : f2bf(0.f);
}

// ---------------------------------------------------------------------------
// Kernel 3: WMMA GEMM  C[M,N] = A[M,K](bf16) * W[N,K](bf16)^T + bias
// MODE 0: +bias          -> fp32 C
// MODE 1: +bias, ReLU    -> bf16 Cb
// MODE 2: +bias +spatial_pos[nv] +pe[t] -> fp32 C AND bf16 Cb   (embed)
// MODE 3: +bias          -> bf16 Cb                              (qkv)
// Block tile 128x64x32, 8 waves; async double-buffered LDS when available.
// ---------------------------------------------------------------------------
template <int MODE>
__global__ __launch_bounds__(256)
void gemm_wmma_kernel(const __bf16* __restrict__ A,
                      const __bf16* __restrict__ W,
                      const float* __restrict__ bias,
                      float* __restrict__ C,
                      __bf16* __restrict__ Cb,
                      int M, int N, int K,
                      const int* __restrict__ meta,
                      const float* __restrict__ spos) {
    __shared__ __align__(16) __bf16 As[2][128][32];
    __shared__ __align__(16) __bf16 Bs[2][64][32];
    const int tid  = threadIdx.x;
    const int wave = tid >> 5, lane = tid & 31;
    const int m16  = lane & 15, half = lane >> 4;
    const int m0 = blockIdx.y * 128, n0 = blockIdx.x * 64;

    f32x8 acc[4] = {};

    auto compute = [&](int buf) {
        bf16x16 af;
        {
            bf16x8 lo = *(const bf16x8*)&As[buf][wave * 16 + m16][half * 8];
            bf16x8 hi = *(const bf16x8*)&As[buf][wave * 16 + m16][16 + half * 8];
#pragma unroll
            for (int e = 0; e < 8; ++e) { af[e] = lo[e]; af[e + 8] = hi[e]; }
        }
#pragma unroll
        for (int nt = 0; nt < 4; ++nt) {
            bf16x16 bfv;
            bf16x8 lo = *(const bf16x8*)&Bs[buf][nt * 16 + m16][half * 8];
            bf16x8 hi = *(const bf16x8*)&Bs[buf][nt * 16 + m16][16 + half * 8];
#pragma unroll
            for (int e = 0; e < 8; ++e) { bfv[e] = lo[e]; bfv[e + 8] = hi[e]; }
            acc[nt] = __builtin_amdgcn_wmma_f32_16x16x32_bf16(
                false, af, false, bfv, (short)0, acc[nt], false, false);
        }
    };

#ifdef USE_ASYNC_LDS
    // each thread issues 3 x 16B async copies per 32-wide K tile
    auto issue = [&](int buf, int k0) {
#pragma unroll
        for (int i = 0; i < 2; ++i) {
            int idx = tid + i * 256;          // 0..511
            int r = idx >> 2, c = (idx & 3) * 8;
            async_cp16(&As[buf][r][c], A + (size_t)(m0 + r) * K + k0 + c);
        }
        {
            int r = tid >> 2, c = (tid & 3) * 8;
            async_cp16(&Bs[buf][r][c], W + (size_t)(n0 + r) * K + k0 + c);
        }
    };
    issue(0, 0);
    int nk = K >> 5;
    for (int it = 0; it < nk; ++it) {
        int cur = it & 1;
        bool more = (it + 1) < nk;
        if (more) issue(cur ^ 1, (it + 1) << 5);
        if (more) async_wait_last3(); else async_wait_all();
        __syncthreads();
        compute(cur);
        __syncthreads();
    }
#else
    for (int k0 = 0; k0 < K; k0 += 32) {
#pragma unroll
        for (int i = 0; i < 2; ++i) {
            int idx = tid + i * 256;
            int r = idx >> 2, c = (idx & 3) * 8;
            *(uint4*)&As[0][r][c] = *(const uint4*)(A + (size_t)(m0 + r) * K + k0 + c);
        }
        {
            int r = tid >> 2, c = (tid & 3) * 8;
            *(uint4*)&Bs[0][r][c] = *(const uint4*)(W + (size_t)(n0 + r) * K + k0 + c);
        }
        if (k0 + 32 < K) {
            __builtin_prefetch(A + (size_t)(m0 + (tid >> 1)) * K + k0 + 32, 0, 0);
            __builtin_prefetch(W + (size_t)(n0 + (tid & 63)) * K + k0 + 32, 0, 0);
        }
        __syncthreads();
        compute(0);
        __syncthreads();
    }
#endif

    // epilogue; C layout: VGPR r -> row (r + half*8), col = lane&15
#pragma unroll
    for (int nt = 0; nt < 4; ++nt) {
#pragma unroll
        for (int r = 0; r < 8; ++r) {
            int row = m0 + wave * 16 + r + half * 8;
            int col = n0 + nt * 16 + m16;
            float v = acc[nt][r] + bias[col];
            if (MODE == 1) v = fmaxf(v, 0.f);
            if (MODE == 2) {
                int tpos = meta[(size_t)row * 2];
                int nv   = meta[(size_t)row * 2 + 1];
                float freq = __expf(-(float)(col & ~1) * (9.2103403719761836f / 256.f));
                float ang  = (float)tpos * freq;
                v += spos[(size_t)nv * DMOD + col] + ((col & 1) ? __cosf(ang) : __sinf(ang));
            }
            if (MODE == 0 || MODE == 2) C[(size_t)row * N + col] = v;
            if (MODE == 1 || MODE == 2 || MODE == 3) Cb[(size_t)row * N + col] = f2bf(v);
        }
    }
}

// ---------------------------------------------------------------------------
// Kernel 4: flash attention, one wave per (b, head, 16-query tile).
// qkv: [B*L, 768] bf16 (q|k|v). out: [B*L, 256] bf16.
// ---------------------------------------------------------------------------
__global__ __launch_bounds__(32)
void attn_kernel(const __bf16* __restrict__ qkv, __bf16* __restrict__ out) {
    const int lane = threadIdx.x;
    const int n = lane & 15, half = lane >> 4;
    const int q0 = blockIdx.x * 16;
    const int h  = blockIdx.y;
    const int b  = blockIdx.z;
    const float scale = 0.17677669529663687f;   // 1/sqrt(32)
    const __bf16 bz = f2bf(0.f);

    __shared__ __bf16 P[16][32];
    for (int i = lane; i < 16 * 16; i += 32) P[i >> 4][16 + (i & 15)] = bz;
    __syncthreads();

    // Q fragment (rows = queries, K-dim = head dims 0..31)
    bf16x16 aq;
#pragma unroll
    for (int e = 0; e < 16; ++e) {
        int d = (e < 8 ? e : e + 8) + half * 8;
        aq[e] = qkv[((size_t)(b * LLEN + q0 + n)) * 768 + h * DHEAD + d];
    }

    float mrow[8], lrow[8];
#pragma unroll
    for (int r = 0; r < 8; ++r) { mrow[r] = -3.0e38f; lrow[r] = 0.f; }
    f32x8 o0 = {}, o1 = {};

    for (int kt = 0; kt < LLEN / 16; ++kt) {
        int k0 = kt * 16;
        // K^T fragment: col n = key, K-dim = head dims
        bf16x16 bk;
#pragma unroll
        for (int e = 0; e < 16; ++e) {
            int d = (e < 8 ? e : e + 8) + half * 8;
            bk[e] = qkv[((size_t)(b * LLEN + k0 + n)) * 768 + 256 + h * DHEAD + d];
        }
        f32x8 s = {};
        s = __builtin_amdgcn_wmma_f32_16x16x32_bf16(false, aq, false, bk, (short)0, s, false, false);

        // online softmax: row r of s lives in lanes sharing `half`; key = lane&15
#pragma unroll
        for (int r = 0; r < 8; ++r) {
            float sv = s[r] * scale;
            float rm = sv;
            rm = fmaxf(rm, __shfl_xor(rm, 8));
            rm = fmaxf(rm, __shfl_xor(rm, 4));
            rm = fmaxf(rm, __shfl_xor(rm, 2));
            rm = fmaxf(rm, __shfl_xor(rm, 1));
            float nm = fmaxf(mrow[r], rm);
            float p  = __expf(sv - nm);
            float ps = p;
            ps += __shfl_xor(ps, 8);
            ps += __shfl_xor(ps, 4);
            ps += __shfl_xor(ps, 2);
            ps += __shfl_xor(ps, 1);
            float corr = __expf(mrow[r] - nm);
            lrow[r] = lrow[r] * corr + ps;
            mrow[r] = nm;
            o0[r] *= corr; o1[r] *= corr;
            P[r + half * 8][n] = f2bf(p);
        }
        __syncthreads();

        // P as A-fragment (keys 16..31 are the zero pad)
        bf16x16 ap;
#pragma unroll
        for (int e = 0; e < 16; ++e) {
            int k = (e < 8 ? e : e + 8) + half * 8;
            ap[e] = P[n][k];
        }
#pragma unroll
        for (int nt = 0; nt < 2; ++nt) {
            bf16x16 bv;
#pragma unroll
            for (int e = 0; e < 16; ++e) {
                int kk = (e < 8 ? e : e + 8) + half * 8;
                bv[e] = (kk < 16)
                    ? qkv[((size_t)(b * LLEN + k0 + kk)) * 768 + 512 + h * DHEAD + nt * 16 + n]
                    : bz;
            }
            if (nt == 0)
                o0 = __builtin_amdgcn_wmma_f32_16x16x32_bf16(false, ap, false, bv, (short)0, o0, false, false);
            else
                o1 = __builtin_amdgcn_wmma_f32_16x16x32_bf16(false, ap, false, bv, (short)0, o1, false, false);
        }
        __syncthreads();
    }

#pragma unroll
    for (int r = 0; r < 8; ++r) {
        int row = q0 + r + half * 8;
        float inv = 1.f / lrow[r];
        out[((size_t)(b * LLEN + row)) * DMOD + h * DHEAD + n]      = f2bf(o0[r] * inv);
        out[((size_t)(b * LLEN + row)) * DMOD + h * DHEAD + 16 + n] = f2bf(o1[r] * inv);
    }
}

// ---------------------------------------------------------------------------
// Kernel 5: out = LayerNorm(x [+ res]) * s + b ; one wave per 256-wide row.
// Optionally emits a bf16 copy for the next GEMM's A operand.
// ---------------------------------------------------------------------------
__global__ __launch_bounds__(256)
void add_ln_kernel(const float* x, const float* res,
                   const float* __restrict__ gs, const float* __restrict__ gb,
                   float* out, __bf16* outb) {
    int wave = threadIdx.x >> 5, lane = threadIdx.x & 31;
    size_t row = (size_t)blockIdx.x * 8 + wave;
    const float* xr = x + row * DMOD;
    float v[8];
    float sum = 0.f;
#pragma unroll
    for (int j = 0; j < 8; ++j) {
        int col = lane + j * 32;
        float t = xr[col];
        if (res) t += res[row * DMOD + col];
        v[j] = t; sum += t;
    }
    sum += __shfl_xor(sum, 16); sum += __shfl_xor(sum, 8);
    sum += __shfl_xor(sum, 4);  sum += __shfl_xor(sum, 2); sum += __shfl_xor(sum, 1);
    float mean = sum * (1.f / 256.f);
    float var = 0.f;
#pragma unroll
    for (int j = 0; j < 8; ++j) { float d = v[j] - mean; var += d * d; }
    var += __shfl_xor(var, 16); var += __shfl_xor(var, 8);
    var += __shfl_xor(var, 4);  var += __shfl_xor(var, 2); var += __shfl_xor(var, 1);
    float inv = rsqrtf(var * (1.f / 256.f) + 1e-5f);
#pragma unroll
    for (int j = 0; j < 8; ++j) {
        int col = lane + j * 32;
        float o = (v[j] - mean) * inv * gs[col] + gb[col];
        out[row * DMOD + col] = o;
        if (outb) outb[row * DMOD + col] = f2bf(o);
    }
}

// ---------------------------------------------------------------------------
extern "C" void kernel_launch(void* const* d_in, const int* in_sizes, int n_in,
                              void* d_out, int out_size, void* d_ws, size_t ws_size,
                              hipStream_t stream) {
    const float*         x_img      = (const float*)d_in[0];
    const unsigned char* patch_mask = (const unsigned char*)d_in[1];
    const int*           valid_idx  = (const int*)d_in[2];
    const float* pe_w  = (const float*)d_in[3];
    const float* pe_b  = (const float*)d_in[4];
    const float* spos  = (const float*)d_in[5];
    const float* Wqkv  = (const float*)d_in[6];
    const float* bqkv  = (const float*)d_in[7];
    const float* Wo    = (const float*)d_in[8];
    const float* bo    = (const float*)d_in[9];
    const float* ln1s  = (const float*)d_in[10];
    const float* ln1b  = (const float*)d_in[11];
    const float* W1    = (const float*)d_in[12];
    const float* b1    = (const float*)d_in[13];
    const float* W2    = (const float*)d_in[14];
    const float* b2    = (const float*)d_in[15];
    const float* ln2s  = (const float*)d_in[16];
    const float* ln2b  = (const float*)d_in[17];
    const float* nfs   = (const float*)d_in[18];
    const float* nfb   = (const float*)d_in[19];

    char* ws = (char*)d_ws;
    size_t off = 0;
    auto alloc = [&](size_t bytes) -> void* {
        void* p = ws + off;
        off += (bytes + 255) & ~(size_t)255;
        return p;
    };
    __bf16* tokA  = (__bf16*)alloc((size_t)MTOT * 128 * sizeof(__bf16));
    int*    meta  = (int*)   alloc((size_t)MTOT * 2 * sizeof(int));
    __bf16* wpadb = (__bf16*)alloc((size_t)DMOD * 128 * sizeof(__bf16));
    __bf16* Wqkvb = (__bf16*)alloc((size_t)NLAY * 768 * DMOD * sizeof(__bf16));
    __bf16* Wob   = (__bf16*)alloc((size_t)NLAY * DMOD * DMOD * sizeof(__bf16));
    __bf16* W1b   = (__bf16*)alloc((size_t)NLAY * DFFN * DMOD * sizeof(__bf16));
    __bf16* W2b   = (__bf16*)alloc((size_t)NLAY * DMOD * DFFN * sizeof(__bf16));
    float*  x     = (float*) alloc((size_t)MTOT * DMOD * sizeof(float));
    float*  tmp   = (float*) alloc((size_t)MTOT * DMOD * sizeof(float));
    __bf16* xbf   = (__bf16*)alloc((size_t)MTOT * DMOD * sizeof(__bf16));
    __bf16* aoutb = (__bf16*)alloc((size_t)MTOT * DMOD * sizeof(__bf16));
    __bf16* big   = (__bf16*)alloc((size_t)MTOT * DFFN * sizeof(__bf16)); // qkv(768)/ffn hidden(1024)

    // 0) one-time weight conversion to bf16
    {
        int n;
        n = NLAY * 768 * DMOD;  cvt_bf16_kernel<<<(n + 255) / 256, 256, 0, stream>>>(Wqkv, Wqkvb, n);
        n = NLAY * DMOD * DMOD; cvt_bf16_kernel<<<(n + 255) / 256, 256, 0, stream>>>(Wo,   Wob,   n);
        n = NLAY * DFFN * DMOD; cvt_bf16_kernel<<<(n + 255) / 256, 256, 0, stream>>>(W1,   W1b,   n);
        n = NLAY * DMOD * DFFN; cvt_bf16_kernel<<<(n + 255) / 256, 256, 0, stream>>>(W2,   W2b,   n);
        pad_w_kernel<<<(DMOD * 128) / 256, 256, 0, stream>>>(pe_w, wpadb);
    }

    // 1) token gather
    build_tokens_kernel<<<BB * TT, 128, 0, stream>>>(x_img, patch_mask, valid_idx, tokA, meta);

    // 2) patch embed + positional encodings: x(fp32) & xbf(bf16)
    gemm_wmma_kernel<2><<<dim3(DMOD / 64, MTOT / 128), 256, 0, stream>>>(
        tokA, wpadb, pe_b, x, xbf, MTOT, DMOD, 128, meta, spos);

    // 3) transformer layers
    for (int i = 0; i < NLAY; ++i) {
        gemm_wmma_kernel<3><<<dim3(768 / 64, MTOT / 128), 256, 0, stream>>>(
            xbf, Wqkvb + (size_t)i * 768 * DMOD, bqkv + (size_t)i * 768, nullptr, big,
            MTOT, 768, DMOD, nullptr, nullptr);
        attn_kernel<<<dim3(LLEN / 16, NHEADS, BB), 32, 0, stream>>>(big, aoutb);
        gemm_wmma_kernel<0><<<dim3(DMOD / 64, MTOT / 128), 256, 0, stream>>>(
            aoutb, Wob + (size_t)i * DMOD * DMOD, bo + (size_t)i * DMOD, tmp, nullptr,
            MTOT, DMOD, DMOD, nullptr, nullptr);
        add_ln_kernel<<<MTOT / 8, 256, 0, stream>>>(
            x, tmp, ln1s + (size_t)i * DMOD, ln1b + (size_t)i * DMOD, x, xbf);
        gemm_wmma_kernel<1><<<dim3(DFFN / 64, MTOT / 128), 256, 0, stream>>>(
            xbf, W1b + (size_t)i * DFFN * DMOD, b1 + (size_t)i * DFFN, nullptr, big,
            MTOT, DFFN, DMOD, nullptr, nullptr);
        gemm_wmma_kernel<0><<<dim3(DMOD / 64, MTOT / 128), 256, 0, stream>>>(
            big, W2b + (size_t)i * DMOD * DFFN, b2 + (size_t)i * DMOD, tmp, nullptr,
            MTOT, DMOD, DFFN, nullptr, nullptr);
        add_ln_kernel<<<MTOT / 8, 256, 0, stream>>>(
            x, tmp, ln2s + (size_t)i * DMOD, ln2b + (size_t)i * DMOD, x, xbf);
    }

    // 4) final layernorm -> d_out (fp32)
    add_ln_kernel<<<MTOT / 8, 256, 0, stream>>>(x, nullptr, nfs, nfb, (float*)d_out, nullptr);
}
